// LearnableMultiHeadSelfAttention_36086315221493
// MI455X (gfx1250) — compile-verified, matrix operand order checked by reading, not run on hardware
//
#include <hip/hip_runtime.h>
#include <hip/hip_bf16.h>

// ---------------- problem constants ----------------
constexpr int CB  = 2;       // batch
constexpr int CL  = 2048;    // sequence length
constexpr int CD  = 1024;    // model dim
constexpr int CH  = 16;      // heads
constexpr int CDH = 64;      // head dim

typedef __attribute__((ext_vector_type(16))) _Float16 v16h;
typedef __attribute__((ext_vector_type(8)))  _Float16 v8h;
typedef __attribute__((ext_vector_type(8)))  float    v8f;

// ---------------- WMMA helpers ----------------
__device__ __forceinline__ v8f wmma_f16(v16h a, v16h b, v8f c) {
  // D = A(16x32 f16) * B(32x16 f16) + C(16x16 f32)
  return __builtin_amdgcn_wmma_f32_16x16x32_f16(
      /*neg_a=*/false, a, /*neg_b=*/false, b,
      /*c_mod=*/(short)0, c, /*reuse_a=*/false, /*reuse_b=*/false);
}

// Load one 16-element K-slice fragment for A (or symmetric B) operand.
// ISA 16-bit A layout: lane<16 holds row M=lane, K = {kb..kb+7, kb+16..kb+23},
// kb = (lane>=16)?8:0. rowptr points at the start of this lane's row.
__device__ __forceinline__ v16h frag_row(const _Float16* rowptr, int k0) {
  int kb = ((threadIdx.x >> 4) & 1) * 8;
  const _Float16* p = rowptr + k0 + kb;
  v8h lo = *(const v8h*)(p);
  v8h hi = *(const v8h*)(p + 16);
  v16h a;
#pragma unroll
  for (int i = 0; i < 8; ++i) { a[i] = lo[i]; a[8 + i] = hi[i]; }
  return a;
}

// ---------------- conversion / layout kernels ----------------
__global__ void cvt_f32_to_f16(const float* __restrict__ src,
                               _Float16* __restrict__ dst, int n) {
  int i = blockIdx.x * blockDim.x + threadIdx.x;
  if (i < n) dst[i] = (_Float16)src[i];
}

// pos_emb [L,1,D] -> reversed rows so row p == relative position p
__global__ void cvt_pos_rev(const float* __restrict__ pe,
                            _Float16* __restrict__ dst) {
  int i = blockIdx.x * blockDim.x + threadIdx.x;
  if (i < CL * CD) {
    int p = i / CD, j = i - p * CD;
    dst[i] = (_Float16)pe[(size_t)(CL - 1 - p) * CD + j];
  }
}

// W [K,N] f32 -> Wt [N,K] f16 (so B operand loads contiguous in K)
__global__ void cvt_w_transpose(const float* __restrict__ W,
                                _Float16* __restrict__ Wt, int K, int N) {
  int i = blockIdx.x * blockDim.x + threadIdx.x;
  if (i < K * N) {
    int n = i / K, k = i - n * K;
    Wt[i] = (_Float16)W[(size_t)k * N + n];
  }
}

// V16 [B,L,H*DH] -> Vt [B,H,DH,L]
__global__ void transpose_v(const _Float16* __restrict__ V16,
                            _Float16* __restrict__ Vt) {
  int i = blockIdx.x * blockDim.x + threadIdx.x;
  if (i < CB * CH * CDH * CL) {
    int l = i % CL;
    int rest = i / CL;
    int d = rest % CDH; rest /= CDH;
    int h = rest % CH;
    int b = rest / CH;
    Vt[i] = V16[((size_t)(b * CL + l)) * CD + h * CDH + d];
  }
}

// ---------------- WMMA GEMM: C = A[MxK] * Bt[NxK]^T + bias ----------------
// Register-blocked: each wave owns a 64x64 output tile (16 accumulators).
// Per K-step(32): 8 fragment loads feed 16 WMMAs -> 4x the WMMA/load
// intensity of a 16x16-tile design; L2 panel traffic drops 4x.
__global__ void __launch_bounds__(128) gemm_wmma(
    const _Float16* __restrict__ A, const _Float16* __restrict__ Bt,
    const float* __restrict__ bias, _Float16* __restrict__ C16,
    float* __restrict__ C32, int M, int N, int K) {
  int lane = threadIdx.x & 31;
  int wid = threadIdx.x >> 5;
  int tile = blockIdx.x * (blockDim.x >> 5) + wid;
  int tn_count = N >> 6;                       // 64-wide N tiles
  int tm = tile / tn_count;
  int tn = tile - tm * tn_count;
  if (tm >= (M >> 6)) return;                  // wave-uniform
  int r = lane & 15;

  const _Float16* arow[4];
  const _Float16* brow[4];
#pragma unroll
  for (int i = 0; i < 4; ++i) {
    arow[i] = A + (size_t)(tm * 64 + 16 * i + r) * K;
    brow[i] = Bt + (size_t)(tn * 64 + 16 * i + r) * K;
  }

  v8f acc[4][4] = {};
  for (int k0 = 0; k0 < K; k0 += 32) {
    __builtin_prefetch(arow[0] + k0 + 1024, 0, 3);   // next A panel -> global_prefetch_b8
    __builtin_prefetch(brow[0] + k0 + 1024, 0, 3);   // next B panel
    v16h af[4], bf[4];
#pragma unroll
    for (int i = 0; i < 4; ++i) af[i] = frag_row(arow[i], k0);
#pragma unroll
    for (int i = 0; i < 4; ++i) bf[i] = frag_row(brow[i], k0);
#pragma unroll
    for (int mi = 0; mi < 4; ++mi)
#pragma unroll
      for (int ni = 0; ni < 4; ++ni)
        acc[mi][ni] = wmma_f16(af[mi], bf[ni], acc[mi][ni]);
  }

  int ro = (lane >> 4) * 8;                    // C layout: rows 0-7 / 8-15
#pragma unroll
  for (int ni = 0; ni < 4; ++ni) {
    int col = tn * 64 + 16 * ni + r;
    float bv = bias[col];
#pragma unroll
    for (int mi = 0; mi < 4; ++mi) {
#pragma unroll
      for (int j = 0; j < 8; ++j) {
        size_t row = (size_t)(tm * 64 + 16 * mi + ro + j);
        float vres = acc[mi][ni][j] + bv;
        if (C16) C16[row * N + col] = (_Float16)vres;
        if (C32) C32[row * N + col] = vres;
      }
    }
  }
}

// ---------------- Transformer-XL flash attention ----------------
// One wave per (b, h, 16-query tile). 32 keys per iteration.
// BD_shifted[q,k] = qv[q] . rrev[q-k]  (rrev = reversed-pos-emb @ Wr)
__global__ void __launch_bounds__(128) txl_attention(
    const _Float16* __restrict__ Q16, const _Float16* __restrict__ K16,
    const _Float16* __restrict__ Vt16, const _Float16* __restrict__ R16,
    const float* __restrict__ ubias, const float* __restrict__ vbias,
    _Float16* __restrict__ ctx16) {
  __shared__ __align__(16) float lds_raw[4][16 * 48];   // 3KB per wave
  const int lane = threadIdx.x & 31;
  const int wid = threadIdx.x >> 5;
  float* g_lds = lds_raw[wid];
  _Float16* p_lds = (_Float16*)lds_raw[wid];

  int gid = blockIdx.x * 4 + wid;
  const int QT = CL / 16;
  int qt = gid % QT;
  int h = (gid / QT) % CH;
  int b = gid / (QT * CH);
  int q0 = qt * 16;

  const int r = lane & 15;
  const int ro = (lane >> 4) * 8;

  // Build A-fragments of qu = q*scale + u and qv = q*scale + v (16x64, two K=32 halves)
  const _Float16* Qrow = Q16 + (size_t)(b * CL + q0 + r) * CD + h * CDH;
  const float* ub = ubias + h * CDH;
  const float* vb = vbias + h * CDH;
  const float scale = 0.125f;                 // 1/sqrt(64)
  v16h qu[2], qv[2];
#pragma unroll
  for (int half = 0; half < 2; ++half) {
    int k0 = 32 * half;
    int kb = ((threadIdx.x >> 4) & 1) * 8;
    v16h au, av;
#pragma unroll
    for (int i = 0; i < 16; ++i) {
      int kk = k0 + kb + (i < 8 ? i : 8 + i);
      float x = (float)Qrow[kk] * scale;
      au[i] = (_Float16)(x + ub[kk]);
      av[i] = (_Float16)(x + vb[kk]);
    }
    qu[half] = au; qv[half] = av;
  }

  v8f o0 = {}, o1 = {}, o2 = {}, o3 = {};
  float rmax[8], rsum[8];
#pragma unroll
  for (int j = 0; j < 8; ++j) { rmax[j] = -3.0e38f; rsum[j] = 0.0f; }

  const _Float16* Kbh = K16 + (size_t)b * CL * CD + h * CDH;
  const _Float16* Vbh = Vt16 + ((size_t)(b * CH + h)) * CDH * CL;
  const _Float16* Rh = R16 + h * CDH;

  const int jend = q0 + 16;
  for (int j0 = 0; j0 < jend; j0 += 32) {
    // ---- AC scores: two 16x16 key tiles, K=64 reduction over dh ----
    v8f s0acc = {}, s1acc = {};
    {
      const _Float16* k0row = Kbh + (size_t)(j0 + r) * CD;
      const _Float16* k1row = Kbh + (size_t)(j0 + 16 + r) * CD;
      s0acc = wmma_f16(qu[0], frag_row(k0row, 0), s0acc);
      s0acc = wmma_f16(qu[1], frag_row(k0row, 32), s0acc);
      s1acc = wmma_f16(qu[0], frag_row(k1row, 0), s1acc);
      s1acc = wmma_f16(qu[1], frag_row(k1row, 32), s1acc);
    }
    // ---- BD band G[qi,n] = qv . rrev[pb+n], n in [0,48), pb = q0-j0-31 ----
    int pb = q0 - j0 - 31;
    asm volatile("s_wait_dscnt 0" ::: "memory");   // WAR vs previous iter's LDS reads
#pragma unroll
    for (int t = 0; t < 3; ++t) {
      int p = pb + 16 * t + r;
      p = p < 0 ? 0 : (p > CL - 1 ? CL - 1 : p);   // clamped rows feed only masked slots
      const _Float16* rrow = Rh + (size_t)p * CD;
      v8f g = {};
      g = wmma_f16(qv[0], frag_row(rrow, 0), g);
      g = wmma_f16(qv[1], frag_row(rrow, 32), g);
#pragma unroll
      for (int j = 0; j < 8; ++j)
        g_lds[(ro + j) * 48 + 16 * t + r] = g[j];
    }
    asm volatile("s_wait_dscnt 0" ::: "memory");

    // ---- gather diagonals, causal mask ----
    float s0v[8], s1v[8];
#pragma unroll
    for (int j = 0; j < 8; ++j) {
      int qi = ro + j;
      int q = q0 + qi;
      float s0 = s0acc[j] + g_lds[qi * 48 + (31 + qi - r)];
      float s1 = s1acc[j] + g_lds[qi * 48 + (15 + qi - r)];
      s0v[j] = (j0 + r <= q) ? s0 : -1e9f;
      s1v[j] = (j0 + 16 + r <= q) ? s1 : -1e9f;
    }
    asm volatile("s_wait_dscnt 0" ::: "memory");   // gathers done before P overwrites

    // ---- online softmax (row reductions across 16-lane halves) ----
#pragma unroll
    for (int j = 0; j < 8; ++j) {
      float tm = fmaxf(s0v[j], s1v[j]);
#pragma unroll
      for (int m = 1; m <= 8; m <<= 1)
        tm = fmaxf(tm, __shfl_xor(tm, m, 32));
      float mnew = fmaxf(rmax[j], tm);
      float corr = __expf(rmax[j] - mnew);
      float p0 = __expf(s0v[j] - mnew);
      float p1 = __expf(s1v[j] - mnew);
      float ts = p0 + p1;
#pragma unroll
      for (int m = 1; m <= 8; m <<= 1)
        ts += __shfl_xor(ts, m, 32);
      rsum[j] = rsum[j] * corr + ts;
      rmax[j] = mnew;
      o0[j] *= corr; o1[j] *= corr; o2[j] *= corr; o3[j] *= corr;
      p_lds[(ro + j) * 32 + r] = (_Float16)p0;
      p_lds[(ro + j) * 32 + 16 + r] = (_Float16)p1;
    }
    asm volatile("s_wait_dscnt 0" ::: "memory");

    // ---- P(16x32) x V(32x64): 4 WMMAs over dh chunks ----
    v16h pa = frag_row(p_lds + r * 32, 0);
    o0 = wmma_f16(pa, frag_row(Vbh + (size_t)(0 + r) * CL + j0, 0), o0);
    o1 = wmma_f16(pa, frag_row(Vbh + (size_t)(16 + r) * CL + j0, 0), o1);
    o2 = wmma_f16(pa, frag_row(Vbh + (size_t)(32 + r) * CL + j0, 0), o2);
    o3 = wmma_f16(pa, frag_row(Vbh + (size_t)(48 + r) * CL + j0, 0), o3);
  }

  // ---- finalize: ctx[b, q, h, dh] = o / rowsum ----
  _Float16* outbase = ctx16 + ((size_t)(b * CL + q0) * CH + h) * CDH;
#pragma unroll
  for (int j = 0; j < 8; ++j) {
    float inv = 1.0f / rsum[j];
    _Float16* pr = outbase + (size_t)(ro + j) * CD;   // q stride = H*DH = D
    pr[0 + r]  = (_Float16)(o0[j] * inv);
    pr[16 + r] = (_Float16)(o1[j] * inv);
    pr[32 + r] = (_Float16)(o2[j] * inv);
    pr[48 + r] = (_Float16)(o3[j] * inv);
  }
}

// ---------------- host-side orchestration ----------------
extern "C" void kernel_launch(void* const* d_in, const int* in_sizes, int n_in,
                              void* d_out, int out_size, void* d_ws, size_t ws_size,
                              hipStream_t stream) {
  (void)in_sizes; (void)n_in; (void)out_size; (void)ws_size;
  const float* x       = (const float*)d_in[0];
  // d_in[1] = bias: causal (1-tril)*-1e9 mask, applied analytically in-kernel
  const float* pos_emb = (const float*)d_in[2];
  const float* u       = (const float*)d_in[3];
  const float* v       = (const float*)d_in[4];
  const float* Wq = (const float*)d_in[5];  const float* bq = (const float*)d_in[6];
  const float* Wk = (const float*)d_in[7];  const float* bk = (const float*)d_in[8];
  const float* Wv = (const float*)d_in[9];  const float* bv = (const float*)d_in[10];
  const float* Wr = (const float*)d_in[11]; const float* br = (const float*)d_in[12];
  const float* Wo = (const float*)d_in[13]; const float* bo = (const float*)d_in[14];
  float* out = (float*)d_out;

  // workspace layout (f16 elements). Total ~58 MB.
  const size_t nBLD = (size_t)CB * CL * CD;   // 4M
  const size_t nLD  = (size_t)CL * CD;        // 2M
  const size_t nDD  = (size_t)CD * CD;        // 1M
  _Float16* p = (_Float16*)d_ws;
  _Float16* x16  = p; p += nBLD;
  _Float16* pe16 = p; p += nLD;
  _Float16* wq16 = p; p += nDD;
  _Float16* wk16 = p; p += nDD;
  _Float16* wv16 = p; p += nDD;
  _Float16* wr16 = p; p += nDD;
  _Float16* wo16 = p; p += nDD;
  _Float16* Q16  = p; p += nBLD;
  _Float16* K16  = p; p += nBLD;
  _Float16* V16  = p; p += nBLD;
  _Float16* R16  = p; p += nLD;
  _Float16* Vt16 = p; p += nBLD;
  _Float16* ctx16 = x16;                       // x16 dead after projections

  const int T = 256;
  cvt_f32_to_f16<<<((int)nBLD + T - 1) / T, T, 0, stream>>>(x, x16, (int)nBLD);
  cvt_pos_rev<<<((int)nLD + T - 1) / T, T, 0, stream>>>(pos_emb, pe16);
  cvt_w_transpose<<<((int)nDD + T - 1) / T, T, 0, stream>>>(Wq, wq16, CD, CD);
  cvt_w_transpose<<<((int)nDD + T - 1) / T, T, 0, stream>>>(Wk, wk16, CD, CD);
  cvt_w_transpose<<<((int)nDD + T - 1) / T, T, 0, stream>>>(Wv, wv16, CD, CD);
  cvt_w_transpose<<<((int)nDD + T - 1) / T, T, 0, stream>>>(Wr, wr16, CD, CD);
  cvt_w_transpose<<<((int)nDD + T - 1) / T, T, 0, stream>>>(Wo, wo16, CD, CD);

  const int M = CB * CL;                            // 4096
  const int tiles_qkv = (M / 64) * (CD / 64);       // 1024 wave-tiles -> 256 blocks
  gemm_wmma<<<tiles_qkv / 4, 128, 0, stream>>>(x16, wq16, bq, Q16, nullptr, M, CD, CD);
  gemm_wmma<<<tiles_qkv / 4, 128, 0, stream>>>(x16, wk16, bk, K16, nullptr, M, CD, CD);
  gemm_wmma<<<tiles_qkv / 4, 128, 0, stream>>>(x16, wv16, bv, V16, nullptr, M, CD, CD);
  const int tiles_r = (CL / 64) * (CD / 64);        // 512 wave-tiles -> 128 blocks
  gemm_wmma<<<tiles_r / 4, 128, 0, stream>>>(pe16, wr16, br, R16, nullptr, CL, CD, CD);

  transpose_v<<<((int)nBLD + T - 1) / T, T, 0, stream>>>(V16, Vt16);

  const int attn_blocks = (CB * CH * (CL / 16)) / 4;   // 1024
  txl_attention<<<attn_blocks, 128, 0, stream>>>(Q16, K16, Vt16, R16, u, v, ctx16);

  gemm_wmma<<<tiles_qkv / 4, 128, 0, stream>>>(ctx16, wo16, bo, nullptr, out, M, CD, CD);
}
